// GraphEncoder_54863912239770
// MI455X (gfx1250) — matrix operand order repaired and checked by reading, not compiled, not run on hardware
//
#include <hip/hip_runtime.h>

typedef unsigned short u16;
typedef __attribute__((ext_vector_type(16))) __bf16 v16bf;
typedef __attribute__((ext_vector_type(8)))  float  v8f;

#if __has_builtin(__builtin_amdgcn_tensor_load_to_lds)
#define HAVE_TDM 1
typedef __attribute__((ext_vector_type(4))) unsigned tdm_v4u;
typedef __attribute__((ext_vector_type(8))) int      tdm_v8i;
typedef __attribute__((ext_vector_type(4))) int      tdm_v4i;
#else
#define HAVE_TDM 0
#endif

#define HEADS 4
#define HID   128
#define LN_EPS 1e-5f
#define NEG_SLOPE 0.2f

// ---------------------------------------------------------------- utilities

__device__ __forceinline__ u16 f2bf(float f) {
  unsigned u = __float_as_uint(f);
  unsigned r = (u + 0x7FFFu + ((u >> 16) & 1u)) >> 16;   // round-to-nearest-even
  return (u16)r;
}

__device__ __forceinline__ void atomicMaxF(float* addr, float v) {
  // works for any sign given init = -inf (0xFF800000)
  if (v >= 0.f) atomicMax((int*)addr, __float_as_int(v));
  else          atomicMin((unsigned int*)addr, __float_as_uint(v));
}

template <int NW>
__device__ __forceinline__ float block_sum(float v, float* red) {
  #pragma unroll
  for (int off = 16; off; off >>= 1) v += __shfl_xor(v, off, 32);
  const int wave = threadIdx.x >> 5, lane = threadIdx.x & 31;
  if (lane == 0) red[wave] = v;
  __syncthreads();
  float t = 0.f;
  #pragma unroll
  for (int i = 0; i < NW; ++i) t += red[i];
  __syncthreads();
  return t;
}

// ---------------------------------------------------------------- kernels

__global__ void cvt_bf16_k(const float* __restrict__ src, u16* __restrict__ dst, int n) {
  int i = blockIdx.x * blockDim.x + threadIdx.x;
  if (i < n) dst[i] = f2bf(src[i]);
}

__global__ void fill_u32_k(unsigned* __restrict__ p, unsigned v, int n) {
  int i = blockIdx.x * blockDim.x + threadIdx.x;
  if (i < n) p[i] = v;
}

// C[M,N] = A[M,K](bf16,row) @ B[K,N](bf16,row), fp32 accumulate.
// Block tile 128(M) x 128(N), K-step 32, 8 wave32 as 4(M) x 2(N): each wave
// owns 32x64 = 8 v_wmma_f32_16x16x32_bf16 per K-step. A tiles are staged by
// the Tensor Data Mover (one wave issues tensor_load_to_lds; TENSORcnt +
// barrier gates consumption; TDM OOB-zero handles the M tail). B tiles are
// staged by vector loads + transposed LDS stores. Both LDS buffers are
// double-buffered so DMA/loads for tile it+1 overlap the WMMA burst on it.
__global__ __launch_bounds__(256)
void gemm_bf16_wmma(const u16* __restrict__ A, const u16* __restrict__ B,
                    float* __restrict__ C, int M, int N, int K) {
  __shared__ u16 As[2][128][32];   // 2 x 8 KB, row-major M x K (TDM destination)
  __shared__ u16 Bs[2][128][32];   // 2 x 8 KB, transposed: [n][k]

  const int tid   = threadIdx.x;
  const int wave  = tid >> 5;
  const int lane  = tid & 31;
  const int waveM = wave & 3;        // 0..3 -> 32-row strip
  const int waveN = wave >> 2;       // 0..1 -> 64-col strip
  const int half  = lane >> 4;       // 0: lanes 0-15, 1: lanes 16-31
  const int lrow  = lane & 15;
  const int mblk  = blockIdx.x * 128;
  const int nblk  = blockIdx.y * 128;

  // B staging assignment (256 threads cover the 8 KB tile)
  const int bk = tid >> 3;           // B k   0..31
  const int bn = (tid & 7) * 16;     // B n   0,16,...,112

  const int NT = K >> 5;
  v8f acc[2][4] = {};

  uint4 b0, b1;                      // register staging for B double buffer
  auto gloadB = [&](int it) {
    const int kk = it << 5;
    const uint4* pb = (const uint4*)(B + (size_t)(kk + bk) * N + nblk + bn);
    b0 = pb[0]; b1 = pb[1];
    if (it + 2 < NT)
      __builtin_prefetch(B + (size_t)((it + 2) * 32 + bk) * N + nblk + bn, 0, 3);
  };
  auto lstoreB = [&](int p) {
    const u16* s0 = (const u16*)&b0;
    const u16* s1 = (const u16*)&b1;
    #pragma unroll
    for (int i = 0; i < 8; ++i) Bs[p][bn + i][bk] = s0[i];
    #pragma unroll
    for (int i = 0; i < 8; ++i) Bs[p][bn + 8 + i][bk] = s1[i];
  };

#if HAVE_TDM
  // Tensor DMA: 2D tile (tile_dim0=32 elems/row, tile_dim1=128 rows) of the
  // bf16 A matrix -> As[p], D# built per CDNA5 ISA.
  const int arows = M - mblk;              // tensor_dim1: OOB rows read as 0
  auto tdm_loadA = [&](int it, int p) {
    if (wave != 0) return;
    unsigned lds = (unsigned)(size_t)&As[p][0][0];          // low 32b = LDS offset
    unsigned long long ga =
        (unsigned long long)(size_t)(A + (size_t)mblk * K + ((size_t)it << 5));
    tdm_v4u g0 = { 1u,                                       // count=1, user mode
                   lds,                                      // lds_addr
                   (unsigned)ga,                             // global_addr[31:0]
                   (unsigned)(ga >> 32) | (2u << 30) };      // addr[56:32] | type=2
    tdm_v8i g1 = { (int)(1u << 16),                          // data_size=1 (2 bytes)
                   (int)((K & 0xFFFF) << 16),                // tensor_dim0[15:0]
                   (int)(((unsigned)K >> 16) | ((arows & 0xFFFF) << 16)),
                   (int)(((unsigned)arows >> 16) | (32u << 16)),  // tile_dim0=32
                   128,                                      // tile_dim1=128 rows
                   K,                                        // dim0_stride[31:0]
                   0, 0 };
    tdm_v4i g2 = { 0, 0, 0, 0 }, g3 = { 0, 0, 0, 0 };
#if __clang_major__ >= 23
    tdm_v8i g4 = { 0, 0, 0, 0, 0, 0, 0, 0 };
    __builtin_amdgcn_tensor_load_to_lds(g0, g1, g2, g3, g4, 0);
#else
    __builtin_amdgcn_tensor_load_to_lds(g0, g1, g2, g3, 0);
#endif
  };
#else
  // fallback: vector-load A staging
  const int ar = tid >> 1;
  const int ac = (tid & 1) * 16;
  uint4 a0, a1;
  auto gloadA = [&](int it) {
    a0 = make_uint4(0u, 0u, 0u, 0u); a1 = a0;
    const int grow = mblk + ar;
    if (grow < M) {
      const uint4* pa = (const uint4*)(A + (size_t)grow * K + (it << 5) + ac);
      a0 = pa[0]; a1 = pa[1];
    }
  };
  auto lstoreA = [&](int p) {
    uint4* da = (uint4*)&As[p][ar][ac];
    da[0] = a0; da[1] = a1;
  };
#endif

  // ---- prologue: stage tile 0
  gloadB(0);
#if HAVE_TDM
  tdm_loadA(0, 0);
#else
  gloadA(0);
  lstoreA(0);
#endif
  lstoreB(0);

  int p = 0;
  union U { v16bf v; uint4 q[2]; };

  for (int it = 0; it < NT; ++it) {
#if HAVE_TDM
    if (wave == 0) __builtin_amdgcn_s_wait_tensorcnt(0);  // As[p] DMA complete
#endif
    __syncthreads();                         // buf[p] ready for everyone
    if (it + 1 < NT) {
      gloadB(it + 1);                        // overlaps the WMMA burst below
#if HAVE_TDM
      tdm_loadA(it + 1, p ^ 1);              // DMA into the idle buffer
#else
      gloadA(it + 1);
#endif
    }

    // --- fragments: distinct registers so all ds_load_b128 pipeline
    U af[2], bfr[4];
    #pragma unroll
    for (int mt = 0; mt < 2; ++mt) {
      // A k order used here (contiguous, same as B): lanes0-15 K0-15, lanes16-31 K16-31
      const uint4* pa = (const uint4*)&As[p][waveM * 32 + mt * 16 + lrow][half ? 16 : 0];
      af[mt].q[0] = pa[0];                   // K 0..7   (or 16..23)
      af[mt].q[1] = pa[1];                   // K 8..15  (or 24..31)
    }
    #pragma unroll
    for (int nt = 0; nt < 4; ++nt) {
      // B k order used here (contiguous, same as A): lanes0-15 hold K0-15 for col=lane, lanes16-31 hold K16-31
      const uint4* pb = (const uint4*)&Bs[p][waveN * 64 + nt * 16 + lrow][half ? 16 : 0];
      bfr[nt].q[0] = pb[0];
      bfr[nt].q[1] = pb[1];
    }
    #pragma unroll
    for (int mt = 0; mt < 2; ++mt)
      #pragma unroll
      for (int nt = 0; nt < 4; ++nt)
        acc[mt][nt] = __builtin_amdgcn_wmma_f32_16x16x32_bf16(
            false, af[mt].v, false, bfr[nt].v, (short)0, acc[mt][nt], false, false);

    __syncthreads();                         // all reads of buf[p] done
    if (it + 1 < NT) {
      lstoreB(p ^ 1);                        // s_wait_loadcnt happens here
#if !HAVE_TDM
      lstoreA(p ^ 1);
#endif
    }
    p ^= 1;
  }

  // --- store C (VGPR i: lanes0-15 -> M=i, lanes16-31 -> M=8+i; N = lane%16)
  if (mblk + 128 <= M) {                     // fast path: whole block in range
    #pragma unroll
    for (int mt = 0; mt < 2; ++mt) {
      #pragma unroll
      for (int nt = 0; nt < 4; ++nt) {
        const int row0 = mblk + waveM * 32 + mt * 16 + half * 8;
        const int col  = nblk + waveN * 64 + nt * 16 + lrow;
        float* cp = C + (size_t)row0 * N + col;
        #pragma unroll
        for (int i = 0; i < 8; ++i) { *cp = acc[mt][nt][i]; cp += N; }
      }
    }
  } else {                                   // M-tail block: per-row guard
    #pragma unroll
    for (int mt = 0; mt < 2; ++mt) {
      #pragma unroll
      for (int nt = 0; nt < 4; ++nt) {
        const int row0 = mblk + waveM * 32 + mt * 16 + half * 8;
        const int col  = nblk + waveN * 64 + nt * 16 + lrow;
        float* cp = C + (size_t)row0 * N + col;
        #pragma unroll
        for (int i = 0; i < 8; ++i) {
          if (row0 + i < M) *cp = acc[mt][nt][i];
          cp += N;
        }
      }
    }
  }
}

// e_src[n,h] = sum_c g[n,h,c]*a_src[h,c]; same for dst. 1 block/node, 1 wave/head.
__global__ __launch_bounds__(128)
void scores_k(const float* __restrict__ g, const float* __restrict__ a_s,
              const float* __restrict__ a_d, float* __restrict__ es,
              float* __restrict__ ed) {
  const int n = blockIdx.x;
  const int h = threadIdx.x >> 5;
  const int lane = threadIdx.x & 31;
  const float* gp  = g   + (size_t)n * (HEADS * HID) + h * HID;
  const float* asp = a_s + h * HID;
  const float* adp = a_d + h * HID;
  float s = 0.f, d = 0.f;
  #pragma unroll
  for (int c = lane; c < HID; c += 32) {
    float v = gp[c];
    s += v * asp[c];
    d += v * adp[c];
  }
  #pragma unroll
  for (int off = 16; off; off >>= 1) { s += __shfl_xor(s, off, 32); d += __shfl_xor(d, off, 32); }
  if (lane == 0) { es[n * HEADS + h] = s; ed[n * HEADS + h] = d; }
}

__device__ __forceinline__ void edge_sd(const int* ei, int NE, int e, int& s, int& d) {
  if (e < NE) { s = ei[e]; d = ei[NE + e]; } else { s = d = e - NE; }  // self loops
}

__global__ void edge_max_k(const int* __restrict__ ei, int NE, int NN,
                           const float* __restrict__ es, const float* __restrict__ ed,
                           float* __restrict__ amax) {
  int e = blockIdx.x * blockDim.x + threadIdx.x;
  if (e >= NE + NN) return;
  int s, d; edge_sd(ei, NE, e, s, d);
  #pragma unroll
  for (int h = 0; h < HEADS; ++h) {
    float a = es[s * HEADS + h] + ed[d * HEADS + h];
    a = (a > 0.f) ? a : NEG_SLOPE * a;
    atomicMaxF(&amax[d * HEADS + h], a);
  }
}

__global__ void edge_exp_k(const int* __restrict__ ei, int NE, int NN,
                           const float* __restrict__ es, const float* __restrict__ ed,
                           const float* __restrict__ amax, float* __restrict__ ex,
                           float* __restrict__ denom) {
  int idx = blockIdx.x * blockDim.x + threadIdx.x;
  if (idx >= (NE + NN) * HEADS) return;
  int e = idx >> 2, h = idx & 3;
  int s, d; edge_sd(ei, NE, e, s, d);
  float a = es[s * HEADS + h] + ed[d * HEADS + h];
  a = (a > 0.f) ? a : NEG_SLOPE * a;
  float x = __expf(a - amax[d * HEADS + h]);
  ex[idx] = x;
  atomicAdd(&denom[d * HEADS + h], x);
}

// one thread per (edge, 4 channels): float4 gather of g[src], 4 scatter-adds
__global__ void edge_agg_k(const int* __restrict__ ei, int NE, int NN,
                           const float* __restrict__ g, const float* __restrict__ ex,
                           const float* __restrict__ denom, float* __restrict__ agg) {
  int idx = blockIdx.x * blockDim.x + threadIdx.x;
  int tot = (NE + NN) * 128;                 // 512 channels / 4
  if (idx >= tot) return;
  int e  = idx >> 7;
  int q  = idx & 127;
  int j4 = q * 4;
  int h  = q >> 5;                           // j4 >> 7
  int s, d; edge_sd(ei, NE, e, s, d);
  float att = ex[e * HEADS + h] / denom[d * HEADS + h];
  float4 gv = *(const float4*)(g + (size_t)s * 512 + j4);
  float* ap = agg + (size_t)d * 512 + j4;
  atomicAdd(ap + 0, gv.x * att);
  atomicAdd(ap + 1, gv.y * att);
  atomicAdd(ap + 2, gv.z * att);
  atomicAdd(ap + 3, gv.w * att);
}

// out+bias -> LayerNorm(512) -> ELU -> bf16
__global__ __launch_bounds__(256)
void ln_elu_bf16_k(const float* __restrict__ agg, const float* __restrict__ bias,
                   const float* __restrict__ gw, const float* __restrict__ bw,
                   u16* __restrict__ out) {
  __shared__ float red[8];
  const int n = blockIdx.x, t = threadIdx.x;
  const float* row = agg + (size_t)n * 512;
  float v0 = row[t] + bias[t];
  float v1 = row[t + 256] + bias[t + 256];
  float m  = block_sum<8>(v0 + v1, red) * (1.f / 512.f);
  float d0 = v0 - m, d1 = v1 - m;
  float var = block_sum<8>(d0 * d0 + d1 * d1, red) * (1.f / 512.f);
  float rs = rsqrtf(var + LN_EPS);
  float y0 = d0 * rs * gw[t] + bw[t];
  float y1 = d1 * rs * gw[t + 256] + bw[t + 256];
  y0 = (y0 > 0.f) ? y0 : __expf(y0) - 1.f;
  y1 = (y1 > 0.f) ? y1 : __expf(y1) - 1.f;
  out[(size_t)n * 512 + t]       = f2bf(y0);
  out[(size_t)n * 512 + t + 256] = f2bf(y1);
}

// mean over heads + bias -> LayerNorm(128) -> bf16
__global__ __launch_bounds__(128)
void mean_ln_bf16_k(const float* __restrict__ agg, const float* __restrict__ bias,
                    const float* __restrict__ gw, const float* __restrict__ bw,
                    u16* __restrict__ out) {
  __shared__ float red[4];
  const int n = blockIdx.x, c = threadIdx.x;
  const float* row = agg + (size_t)n * 512;
  float v = 0.25f * (row[c] + row[128 + c] + row[256 + c] + row[384 + c]) + bias[c];
  float m  = block_sum<4>(v, red) * (1.f / 128.f);
  float d  = v - m;
  float var = block_sum<4>(d * d, red) * (1.f / 128.f);
  float y = d * rsqrtf(var + LN_EPS) * gw[c] + bw[c];
  out[(size_t)n * 128 + c] = f2bf(y);
}

// proj + bias -> LayerNorm(128) -> fp32 output
__global__ __launch_bounds__(128)
void bias_ln_out_k(const float* __restrict__ g, const float* __restrict__ bias,
                   const float* __restrict__ gw, const float* __restrict__ bw,
                   float* __restrict__ out) {
  __shared__ float red[4];
  const int n = blockIdx.x, c = threadIdx.x;
  float v = g[(size_t)n * 128 + c] + bias[c];
  float m  = block_sum<4>(v, red) * (1.f / 128.f);
  float d  = v - m;
  float var = block_sum<4>(d * d, red) * (1.f / 128.f);
  out[(size_t)n * 128 + c] = d * rsqrtf(var + LN_EPS) * gw[c] + bw[c];
}

// ---------------------------------------------------------------- host side

static void run_gat_layer(hipStream_t stream, const u16* hb, int K, const float* W,
                          const float* a_s, const float* a_d, const int* ei, int NE,
                          int NN, u16* wb, float* g, float* agg, float* es, float* ed,
                          float* amax, float* denom, float* ex) {
  const int N = HEADS * HID;           // 512
  const int ET = NE + NN;
  cvt_bf16_k<<<(K * N + 255) / 256, 256, 0, stream>>>(W, wb, K * N);
  dim3 gg((NN + 127) / 128, N / 128);
  gemm_bf16_wmma<<<gg, 256, 0, stream>>>(hb, wb, g, NN, N, K);
  scores_k<<<NN, 128, 0, stream>>>(g, a_s, a_d, es, ed);
  fill_u32_k<<<(NN * HEADS + 255) / 256, 256, 0, stream>>>((unsigned*)amax, 0xFF800000u, NN * HEADS);
  fill_u32_k<<<(NN * HEADS + 255) / 256, 256, 0, stream>>>((unsigned*)denom, 0u, NN * HEADS);
  fill_u32_k<<<(NN * N + 255) / 256, 256, 0, stream>>>((unsigned*)agg, 0u, NN * N);
  edge_max_k<<<(ET + 255) / 256, 256, 0, stream>>>(ei, NE, NN, es, ed, amax);
  edge_exp_k<<<(ET * HEADS + 255) / 256, 256, 0, stream>>>(ei, NE, NN, es, ed, amax, ex, denom);
  int tot = ET * 128;                  // (edge, channel/4) work items
  edge_agg_k<<<(tot + 255) / 256, 256, 0, stream>>>(ei, NE, NN, g, ex, denom, agg);
}

extern "C" void kernel_launch(void* const* d_in, const int* in_sizes, int n_in,
                              void* d_out, int out_size, void* d_ws, size_t ws_size,
                              hipStream_t stream) {
  const float* x      = (const float*)d_in[0];
  const int*   ei     = (const int*)d_in[1];   // [2, NE] int32
  const float* W1     = (const float*)d_in[3];
  const float* a_src1 = (const float*)d_in[4];
  const float* a_dst1 = (const float*)d_in[5];
  const float* b1     = (const float*)d_in[6];
  const float* ln1_g  = (const float*)d_in[7];
  const float* ln1_b  = (const float*)d_in[8];
  const float* W2     = (const float*)d_in[9];
  const float* a_src2 = (const float*)d_in[10];
  const float* a_dst2 = (const float*)d_in[11];
  const float* b2     = (const float*)d_in[12];
  const float* ln2_g  = (const float*)d_in[13];
  const float* ln2_b  = (const float*)d_in[14];
  const float* projW  = (const float*)d_in[15];
  const float* projb  = (const float*)d_in[16];
  const float* on_g   = (const float*)d_in[17];
  const float* on_b   = (const float*)d_in[18];

  const int NN = in_sizes[0] / 256;    // 20000 nodes
  const int NE = in_sizes[1] / 2;      // 320000 edges
  const int ET = NE + NN;

  char* ws = (char*)d_ws;
  size_t off = 0;
  auto take = [&](size_t bytes) -> void* {
    void* p = ws + off;
    off = (off + bytes + 255) & ~(size_t)255;
    return p;
  };
  u16*   xb    = (u16*)take((size_t)NN * 512 * sizeof(u16));  // bf16 activations (reused)
  u16*   wb    = (u16*)take((size_t)512 * 512 * sizeof(u16)); // bf16 weights (reused)
  float* g     = (float*)take((size_t)NN * 512 * sizeof(float));
  float* agg   = (float*)take((size_t)NN * 512 * sizeof(float));
  float* es    = (float*)take((size_t)NN * HEADS * sizeof(float));
  float* ed    = (float*)take((size_t)NN * HEADS * sizeof(float));
  float* amax  = (float*)take((size_t)NN * HEADS * sizeof(float));
  float* denom = (float*)take((size_t)NN * HEADS * sizeof(float));
  float* ex    = (float*)take((size_t)ET * HEADS * sizeof(float));

  // ---- layer 1: x (fp32) -> bf16, GAT(256 -> 4x128 concat), LN, ELU -> bf16
  cvt_bf16_k<<<(NN * 256 + 255) / 256, 256, 0, stream>>>(x, xb, NN * 256);
  run_gat_layer(stream, xb, 256, W1, a_src1, a_dst1, ei, NE, NN,
                wb, g, agg, es, ed, amax, denom, ex);
  ln_elu_bf16_k<<<NN, 256, 0, stream>>>(agg, b1, ln1_g, ln1_b, xb);

  // ---- layer 2: GAT(512 -> 4x128 mean), LN -> bf16
  run_gat_layer(stream, xb, 512, W2, a_src2, a_dst2, ei, NE, NN,
                wb, g, agg, es, ed, amax, denom, ex);
  mean_ln_bf16_k<<<NN, 128, 0, stream>>>(agg, b2, ln2_g, ln2_b, xb);

  // ---- projection 128x128 + LN -> output (fp32)
  cvt_bf16_k<<<(128 * 128 + 255) / 256, 256, 0, stream>>>(projW, wb, 128 * 128);
  dim3 pg((NN + 127) / 128, 1);
  gemm_bf16_wmma<<<pg, 256, 0, stream>>>(xb, wb, g, NN, 128, 128);
  bias_ln_out_k<<<NN, 128, 0, stream>>>(g, projb, on_g, on_b, (float*)d_out);
}
